// HPIGNN_36215164240487
// MI455X (gfx1250) — compile-verified, matrix operand order checked by reading, not run on hardware
//
#include <hip/hip_runtime.h>
#include <hip/hip_bf16.h>
#include <math.h>

// ---------------------------------------------------------------------------
// HGT GNN pipeline for MI455X (gfx1250, wave32).
// Dense GEMMs via v_wmma_f32_16x16x32_f16 (f32 accum) with pre-packed f16
// B-fragments (vector loads, no per-element cvt); edge attention via L2
// atomics (segment softmax + scatter-add).  All launches on `stream`.
// ---------------------------------------------------------------------------

typedef __attribute__((ext_vector_type(16))) _Float16 v16h;
typedef __attribute__((ext_vector_type(8)))  float    v8f;

#define NBUS 100000
#define NGEN 50000
#define NGMD 50000
#define HID  64

__device__ __forceinline__ float gelu_f(float x) {
    return 0.5f * x * (1.0f + erff(x * 0.70710678118654752f));
}

// monotone float<->uint encoding for atomicMax on floats (handles negatives)
__device__ __forceinline__ unsigned fkey(float f) {
    unsigned u = __float_as_uint(f);
    return (u & 0x80000000u) ? ~u : (u | 0x80000000u);
}
__device__ __forceinline__ float funkey(unsigned u) {
    return (u & 0x80000000u) ? __uint_as_float(u & 0x7fffffffu)
                             : __uint_as_float(~u);
}

// ---------------------------------------------------------------------------
// Weight pre-pack: W[K,64] f32 -> B fragments f16, laid out so that lane L of
// a wave reads its 16 f16 operand halves as one contiguous 32-byte vector.
// Fragment order: [ks][t][lane][16], ks = k-step (32 per step), t = 16-col tile.
// B-matrix 32x16 f16 layout (ISA 7.12.2): lane L holds col (L&15); lanes<16
// carry K = {0..7,16..23}+k0, lanes>=16 carry K = {8..15,24..31}+k0.
// ---------------------------------------------------------------------------
__global__ void pack_w_k(const float* __restrict__ W, _Float16* __restrict__ F,
                         int K)
{
    int tid = blockIdx.x * 256 + threadIdx.x;
    int nks = (K + 31) / 32;
    if (tid >= nks * 128) return;
    int lane = tid & 31;
    int t    = (tid >> 5) & 3;
    int k0   = (tid >> 7) * 32;
    int rsel = lane & 15;
    int klo  = (lane & 16) ? 8 : 0;
    int col  = t * 16 + rsel;
    _Float16* o = F + (size_t)tid * 16;
#pragma unroll
    for (int i = 0; i < 8; ++i) {
        int ka = k0 + klo + i;
        int kb = k0 + klo + 16 + i;
        o[i]   = (_Float16)((ka < K) ? W[ka * 64 + col] : 0.0f);
        o[i+8] = (_Float16)((kb < K) ? W[kb * 64 + col] : 0.0f);
    }
}

// Pre-pack per-relation head transform R[2,32,32] into 4 tile fragments
// (t = head*2 + colhalf), single k-step each.
__global__ void pack_r_k(const float* __restrict__ R, _Float16* __restrict__ F)
{
    int tid = threadIdx.x;
    if (tid >= 128) return;
    int lane = tid & 31;
    int t    = tid >> 5;
    int head = t >> 1;
    int rsel = lane & 15;
    int klo  = (lane & 16) ? 8 : 0;
    int coll = (t & 1) * 16 + rsel;
    const float* Rh = R + head * 1024;
    _Float16* o = F + (size_t)tid * 16;
#pragma unroll
    for (int i = 0; i < 8; ++i) {
        o[i]   = (_Float16)Rh[(klo + i) * 32 + coll];
        o[i+8] = (_Float16)Rh[(klo + 16 + i) * 32 + coll];
    }
}

// ---------------------------------------------------------------------------
// WMMA GEMM: out[M,64] = epilogue(A[M,K] @ W[K,64] + bias), W pre-packed f16.
// One wave = one 16-row x 64-col tile (4 accumulators), K looped by 32.
// A fragments loaded as 4x float4 (two 8-float contiguous runs per lane).
// IN_GELU: exact GELU applied to A elements while loading.
// EPI: 0 = bias only; 1 = bias+ReLU; 2 = sigmoid-skip mix with hprev + ReLU.
// ---------------------------------------------------------------------------
template<int IN_GELU, int EPI>
__global__ void gemm_wmma_k(const float* __restrict__ A,
                            const _Float16* __restrict__ F,
                            const float* __restrict__ bias,
                            const float* __restrict__ hprev,
                            const float* __restrict__ skipp,
                            float* __restrict__ out, int M, int K)
{
    int lane = threadIdx.x & 31;
    int wave = threadIdx.x >> 5;
    int tile = blockIdx.x * 8 + wave;
    if (tile * 16 >= M) return;              // wave-uniform: EXEC stays all-1s
    int row0 = tile * 16;
    int rsel = lane & 15;
    int klo  = (lane & 16) ? 8 : 0;
    const float* ar = A + (size_t)(row0 + rsel) * K;

    v8f c0 = {0,0,0,0,0,0,0,0}, c1 = c0, c2 = c0, c3 = c0;
    const float4 z4 = {0.f, 0.f, 0.f, 0.f};

    for (int k0 = 0; k0 < K; k0 += 32) {
        int kb = k0 + klo;
        float4 p0 = (kb + 3  < K) ? *(const float4*)(ar + kb)      : z4;
        float4 p1 = (kb + 7  < K) ? *(const float4*)(ar + kb + 4)  : z4;
        float4 p2 = (kb + 19 < K) ? *(const float4*)(ar + kb + 16) : z4;
        float4 p3 = (kb + 23 < K) ? *(const float4*)(ar + kb + 20) : z4;
        if (IN_GELU) {
            p0.x = gelu_f(p0.x); p0.y = gelu_f(p0.y); p0.z = gelu_f(p0.z); p0.w = gelu_f(p0.w);
            p1.x = gelu_f(p1.x); p1.y = gelu_f(p1.y); p1.z = gelu_f(p1.z); p1.w = gelu_f(p1.w);
            p2.x = gelu_f(p2.x); p2.y = gelu_f(p2.y); p2.z = gelu_f(p2.z); p2.w = gelu_f(p2.w);
            p3.x = gelu_f(p3.x); p3.y = gelu_f(p3.y); p3.z = gelu_f(p3.z); p3.w = gelu_f(p3.w);
        }
        v16h a;
        a[0]  = (_Float16)p0.x; a[1]  = (_Float16)p0.y;
        a[2]  = (_Float16)p0.z; a[3]  = (_Float16)p0.w;
        a[4]  = (_Float16)p1.x; a[5]  = (_Float16)p1.y;
        a[6]  = (_Float16)p1.z; a[7]  = (_Float16)p1.w;
        a[8]  = (_Float16)p2.x; a[9]  = (_Float16)p2.y;
        a[10] = (_Float16)p2.z; a[11] = (_Float16)p2.w;
        a[12] = (_Float16)p3.x; a[13] = (_Float16)p3.y;
        a[14] = (_Float16)p3.z; a[15] = (_Float16)p3.w;

        const v16h* bf = (const v16h*)(F + (size_t)((k0 >> 5) * 128 + lane) * 16);
#pragma unroll
        for (int t = 0; t < 4; ++t) {
            v16h b = bf[t * 32];             // [ks][t][lane][16]
            v8f* cp = (t == 0) ? &c0 : (t == 1) ? &c1 : (t == 2) ? &c2 : &c3;
            *cp = __builtin_amdgcn_wmma_f32_16x16x32_f16(
                      false, a, false, b, (short)0, *cp, false, false);
        }
    }

    float sk = 0.f, osk = 0.f;
    if (EPI == 2) {
        float s = skipp[0];
        sk = 1.0f / (1.0f + __expf(-s));
        osk = 1.0f - sk;
    }
    int rofs = (lane & 16) ? 8 : 0;
#pragma unroll
    for (int t = 0; t < 4; ++t) {
        int col = t * 16 + rsel;
        float bv = bias[col];
        v8f cc = (t == 0) ? c0 : (t == 1) ? c1 : (t == 2) ? c2 : c3;
#pragma unroll
        for (int r = 0; r < 8; ++r) {
            int mrow = row0 + rofs + r;
            float v = cc[r] + bv;
            if (EPI == 2) {
                v = sk * v + osk * hprev[(size_t)mrow * 64 + col];
                v = fmaxf(v, 0.0f);
            } else if (EPI == 1) {
                v = fmaxf(v, 0.0f);
            }
            out[(size_t)mrow * 64 + col] = v;
        }
    }
}

// ---------------------------------------------------------------------------
// Per-relation head transform: out[n, h*32+e] = sum_d X[n, h*32+d]*R[h][d][e]
// Block-diagonal GEMM (D=32 = one k-step per head), R pre-packed fragments.
// ---------------------------------------------------------------------------
__global__ void rel_transform_k(const float* __restrict__ X,
                                const _Float16* __restrict__ F,
                                float* __restrict__ out, int M)
{
    int lane = threadIdx.x & 31;
    int wave = threadIdx.x >> 5;
    int tile = blockIdx.x * 8 + wave;
    if (tile * 16 >= M) return;
    int row0 = tile * 16;
    int rsel = lane & 15;
    int klo  = (lane & 16) ? 8 : 0;
    const float* xr = X + (size_t)(row0 + rsel) * 64 + klo;

    float4 q0 = *(const float4*)(xr);
    float4 q1 = *(const float4*)(xr + 4);
    float4 q2 = *(const float4*)(xr + 16);
    float4 q3 = *(const float4*)(xr + 20);
    float4 q4 = *(const float4*)(xr + 32);
    float4 q5 = *(const float4*)(xr + 36);
    float4 q6 = *(const float4*)(xr + 48);
    float4 q7 = *(const float4*)(xr + 52);

    v16h a0, a1;
    a0[0]  = (_Float16)q0.x; a0[1]  = (_Float16)q0.y;
    a0[2]  = (_Float16)q0.z; a0[3]  = (_Float16)q0.w;
    a0[4]  = (_Float16)q1.x; a0[5]  = (_Float16)q1.y;
    a0[6]  = (_Float16)q1.z; a0[7]  = (_Float16)q1.w;
    a0[8]  = (_Float16)q2.x; a0[9]  = (_Float16)q2.y;
    a0[10] = (_Float16)q2.z; a0[11] = (_Float16)q2.w;
    a0[12] = (_Float16)q3.x; a0[13] = (_Float16)q3.y;
    a0[14] = (_Float16)q3.z; a0[15] = (_Float16)q3.w;
    a1[0]  = (_Float16)q4.x; a1[1]  = (_Float16)q4.y;
    a1[2]  = (_Float16)q4.z; a1[3]  = (_Float16)q4.w;
    a1[4]  = (_Float16)q5.x; a1[5]  = (_Float16)q5.y;
    a1[6]  = (_Float16)q5.z; a1[7]  = (_Float16)q5.w;
    a1[8]  = (_Float16)q6.x; a1[9]  = (_Float16)q6.y;
    a1[10] = (_Float16)q6.z; a1[11] = (_Float16)q6.w;
    a1[12] = (_Float16)q7.x; a1[13] = (_Float16)q7.y;
    a1[14] = (_Float16)q7.z; a1[15] = (_Float16)q7.w;

    int rofs = (lane & 16) ? 8 : 0;
    const v16h* bf = (const v16h*)(F + (size_t)lane * 16);
#pragma unroll
    for (int t = 0; t < 4; ++t) {
        v16h b = bf[t * 32];                 // [t][lane][16]
        v8f acc = {0,0,0,0,0,0,0,0};
        acc = __builtin_amdgcn_wmma_f32_16x16x32_f16(
                  false, (t < 2) ? a0 : a1, false, b, (short)0, acc,
                  false, false);
#pragma unroll
        for (int r = 0; r < 8; ++r)
            out[(size_t)(row0 + rofs + r) * 64 + t * 16 + rsel] = acc[r];
    }
}

// ---------------------------------------------------------------------------
// Edge kernels (attention): gather dot -> seg-max -> exp/seg-sum -> scatter
// ei layout: [2,E] row-major -> src = ei[e], dst = ei[E+e]
// ---------------------------------------------------------------------------
__global__ void edge_score_k(const int* __restrict__ ei, int E,
                             const float* __restrict__ q,
                             const float* __restrict__ krel,
                             const float* __restrict__ prel,
                             float* __restrict__ score,
                             unsigned* __restrict__ segmax)
{
    int e = blockIdx.x * 256 + threadIdx.x;
    if (e >= E) return;
    int s = ei[e], d = ei[E + e];
    const float4* qv = (const float4*)(q + (size_t)d * 64);
    const float4* kv = (const float4*)(krel + (size_t)s * 64);
    float acc0 = 0.f, acc1 = 0.f;
#pragma unroll
    for (int j = 0; j < 8; ++j) {
        float4 a = qv[j], b = kv[j];
        acc0 += a.x * b.x + a.y * b.y + a.z * b.z + a.w * b.w;
    }
#pragma unroll
    for (int j = 8; j < 16; ++j) {
        float4 a = qv[j], b = kv[j];
        acc1 += a.x * b.x + a.y * b.y + a.z * b.z + a.w * b.w;
    }
    const float inv_sqrt_d = 0.17677669529663689f;   // 1/sqrt(32)
    acc0 *= prel[0] * inv_sqrt_d;
    acc1 *= prel[1] * inv_sqrt_d;
    score[(size_t)e * 2]     = acc0;
    score[(size_t)e * 2 + 1] = acc1;
    atomicMax(&segmax[(size_t)d * 2],     fkey(acc0));
    atomicMax(&segmax[(size_t)d * 2 + 1], fkey(acc1));
}

__global__ void edge_expsum_k(const int* __restrict__ ei, int E,
                              float* __restrict__ score,
                              const unsigned* __restrict__ segmax,
                              float* __restrict__ segsum)
{
    int t = blockIdx.x * 256 + threadIdx.x;
    if (t >= E * 2) return;
    int e = t >> 1, h = t & 1;
    int d = ei[E + e];
    float m = funkey(segmax[(size_t)d * 2 + h]);
    if (!isfinite(m)) m = 0.0f;                 // PyG: non-finite max -> 0
    float ex = __expf(score[t] - m);
    score[t] = ex;
    atomicAdd(&segsum[(size_t)d * 2 + h], ex);
}

__global__ void edge_message_k(const int* __restrict__ ei, int E,
                               const float* __restrict__ score,
                               const float* __restrict__ segsum,
                               const float* __restrict__ vrel,
                               float* __restrict__ acc)
{
    int t = blockIdx.x * 256 + threadIdx.x;
    if (t >= E * 4) return;
    int e = t >> 2, cb = t & 3, h = cb >> 1;
    int s = ei[e], d = ei[E + e];
    float alpha = score[(size_t)e * 2 + h] /
                  (segsum[(size_t)d * 2 + h] + 1e-16f);
    const float4* vv = (const float4*)(vrel + (size_t)s * 64 + cb * 16);
    float* ap = acc + (size_t)d * 64 + cb * 16;
#pragma unroll
    for (int j = 0; j < 4; ++j) {
        float4 x = vv[j];
        atomicAdd(ap + j * 4 + 0, x.x * alpha);
        atomicAdd(ap + j * 4 + 1, x.y * alpha);
        atomicAdd(ap + j * 4 + 2, x.z * alpha);
        atomicAdd(ap + j * 4 + 3, x.w * alpha);
    }
}

// ---------------------------------------------------------------------------
// elementwise helpers
// ---------------------------------------------------------------------------
__global__ void fill_f32_k(float* p, size_t n, float v) {
    size_t i = (size_t)blockIdx.x * 256 + threadIdx.x;
    size_t st = (size_t)gridDim.x * 256;
    for (; i < n; i += st) p[i] = v;
}
__global__ void fill_u32_k(unsigned* p, size_t n, unsigned v) {
    size_t i = (size_t)blockIdx.x * 256 + threadIdx.x;
    size_t st = (size_t)gridDim.x * 256;
    for (; i < n; i += st) p[i] = v;
}
__global__ void emin_k(float* dst, const float* src, size_t n) {
    size_t i = (size_t)blockIdx.x * 256 + threadIdx.x;
    if (i < n) dst[i] = fminf(dst[i], src[i]);
}

// ---------------------------------------------------------------------------
// Batch norm (train-mode, biased var): column sums via block-reduced atomics
// ---------------------------------------------------------------------------
__global__ void bn_stats_k(const float* __restrict__ x,
                           float* __restrict__ S, float* __restrict__ Q, int M)
{
    __shared__ float ls[256], lq[256];
    int col = threadIdx.x & 63;
    int rg  = threadIdx.x >> 6;          // 0..3
    float s = 0.f, q = 0.f;
    for (int r = blockIdx.x * 4 + rg; r < M; r += gridDim.x * 4) {
        float v = x[(size_t)r * 64 + col];
        s += v; q += v * v;
    }
    ls[threadIdx.x] = s; lq[threadIdx.x] = q;
    __syncthreads();
    if (threadIdx.x < 64) {
        float Sv = ls[threadIdx.x] + ls[threadIdx.x + 64] +
                   ls[threadIdx.x + 128] + ls[threadIdx.x + 192];
        float Qv = lq[threadIdx.x] + lq[threadIdx.x + 64] +
                   lq[threadIdx.x + 128] + lq[threadIdx.x + 192];
        atomicAdd(&S[threadIdx.x], Sv);
        atomicAdd(&Q[threadIdx.x], Qv);
    }
}

__global__ void bn_apply_k(float* __restrict__ x,
                           const float* __restrict__ S,
                           const float* __restrict__ Q,
                           const float* __restrict__ g,
                           const float* __restrict__ be, int M)
{
    int i = blockIdx.x * 256 + threadIdx.x;
    int n = M * 64;
    if (i >= n) return;
    int c = i & 63;
    float mu  = S[c] / (float)M;
    float var = Q[c] / (float)M - mu * mu;
    float v = (x[i] - mu) * rsqrtf(var + 1e-5f) * g[c] + be[c];
    x[i] = fmaxf(v, 0.0f);
}

__global__ void final_dot_k(const float* __restrict__ x,
                            const float* __restrict__ w3,
                            const float* __restrict__ b3,
                            float* __restrict__ out, int M)
{
    int i = blockIdx.x * 256 + threadIdx.x;
    if (i >= M) return;
    const float4* xr = (const float4*)(x + (size_t)i * 64);
    float acc = 0.f;
#pragma unroll
    for (int j = 0; j < 16; ++j) {
        float4 a = xr[j];
        acc += a.x * w3[j*4] + a.y * w3[j*4+1] + a.z * w3[j*4+2] + a.w * w3[j*4+3];
    }
    out[i] = acc + b3[0];
}

// ---------------------------------------------------------------------------
// host orchestration
// ---------------------------------------------------------------------------
struct ConvP {
    const float *kw[3], *kb[3], *qw[3], *qb[3], *vw[3], *vb[3], *aw[3], *ab[3];
    const float *skip[3];
    const float *arel[5], *mrel[5], *prel[5];
};

extern "C" void kernel_launch(void* const* d_in, const int* in_sizes, int n_in,
                              void* d_out, int out_size, void* d_ws, size_t ws_size,
                              hipStream_t stream)
{
    (void)in_sizes; (void)n_in; (void)out_size; (void)ws_size;
    const int NT[3] = { NBUS, NGEN, NGMD };

    // ---- unpack inputs (setup_inputs() insertion order) ----
    int ix = 0;
    const float* x_in[3];
    x_in[0] = (const float*)d_in[ix++];           // x_bus [100000,32]
    x_in[1] = (const float*)d_in[ix++];           // x_gen [50000,16]
    x_in[2] = (const float*)d_in[ix++];           // x_gmd [50000,16]
    const int* ei[5];
    for (int r = 0; r < 5; ++r) ei[r] = (const int*)d_in[ix++]; // bb,gb,mb,bg,bm

    const float *lin_w[3], *lin_b[3];
    for (int t = 0; t < 3; ++t) {
        lin_w[t] = (const float*)d_in[ix++];
        lin_b[t] = (const float*)d_in[ix++];
    }
    ConvP cv[2];
    for (int l = 0; l < 2; ++l) {
        const float** mats[8] = { cv[l].kw, cv[l].kb, cv[l].qw, cv[l].qb,
                                  cv[l].vw, cv[l].vb, cv[l].aw, cv[l].ab };
        for (int m = 0; m < 8; ++m)
            for (int t = 0; t < 3; ++t) mats[m][t] = (const float*)d_in[ix++];
        for (int t = 0; t < 3; ++t) cv[l].skip[t] = (const float*)d_in[ix++];
        for (int r = 0; r < 5; ++r) cv[l].arel[r] = (const float*)d_in[ix++];
        for (int r = 0; r < 5; ++r) cv[l].mrel[r] = (const float*)d_in[ix++];
        for (int r = 0; r < 5; ++r) cv[l].prel[r] = (const float*)d_in[ix++];
    }
    const float* mw1 = (const float*)d_in[ix++];
    const float* mb1 = (const float*)d_in[ix++];
    const float* mg1 = (const float*)d_in[ix++];
    const float* me1 = (const float*)d_in[ix++];
    const float* mw2 = (const float*)d_in[ix++];
    const float* mb2 = (const float*)d_in[ix++];
    const float* mg2 = (const float*)d_in[ix++];
    const float* me2 = (const float*)d_in[ix++];
    const float* mw3 = (const float*)d_in[ix++];
    const float* mb3 = (const float*)d_in[ix++];

    // ---- workspace layout (~395 MB of f32) ----
    float* base = (float*)d_ws;
    size_t p = 0;
    auto take = [&](size_t n) { float* r = base + p; p += n; return r; };
    float* h0[3]; for (int t = 0; t < 3; ++t) h0[t] = take((size_t)NT[t] * 64);
    float* h1[3]; for (int t = 0; t < 3; ++t) h1[t] = take((size_t)NT[t] * 64);
    float* kf[3]; for (int t = 0; t < 3; ++t) kf[t] = take((size_t)NT[t] * 64);
    float* qf[3]; for (int t = 0; t < 3; ++t) qf[t] = take((size_t)NT[t] * 64);
    float* vf[3]; for (int t = 0; t < 3; ++t) vf[t] = take((size_t)NT[t] * 64);
    float* krel   = take((size_t)NBUS * 64);
    float* vrel   = take((size_t)NBUS * 64);
    float* score  = take((size_t)2000000);           // E_max * H
    unsigned* segmax = (unsigned*)take((size_t)NBUS * 2);
    float* segsum = take((size_t)NBUS * 2);
    float* accA[3]; for (int t = 0; t < 3; ++t) accA[t] = take((size_t)NT[t] * 64);
    float* acctmp = take((size_t)NBUS * 64);
    float* bnS = take(64);
    float* bnQ = take(64);
    _Float16* wpack = (_Float16*)take(2048);         // 2 ksteps * 4 t * 32 * 16 f16
    _Float16* rpack = (_Float16*)take(1024);         // 4 t * 32 * 16 f16
    float* t1 = acctmp;                              // MLP scratch reuse
    float* t2 = krel;

    auto gemm = [&](const float* A, const float* Wt, const float* bias,
                    float* out, int M, int K, int mode,
                    const float* hprev = nullptr, const float* skipp = nullptr) {
        pack_w_k<<<1, 256, 0, stream>>>(Wt, wpack, K);
        int grid = (M / 16 + 7) / 8;
        if (mode == 0)
            gemm_wmma_k<0,0><<<grid,256,0,stream>>>(A,wpack,bias,nullptr,nullptr,out,M,K);
        else if (mode == 1)
            gemm_wmma_k<0,1><<<grid,256,0,stream>>>(A,wpack,bias,nullptr,nullptr,out,M,K);
        else
            gemm_wmma_k<1,2><<<grid,256,0,stream>>>(A,wpack,bias,hprev,skipp,out,M,K);
    };
    auto reltf = [&](const float* X, const float* R, float* out, int M) {
        pack_r_k<<<1, 128, 0, stream>>>(R, rpack);
        int grid = (M / 16 + 7) / 8;
        rel_transform_k<<<grid,256,0,stream>>>(X, rpack, out, M);
    };
    auto zerof = [&](float* ptr, size_t n) {
        int g = (int)((n + 255) / 256); if (g > 2048) g = 2048;
        fill_f32_k<<<g,256,0,stream>>>(ptr, n, 0.0f);
    };

    // ---- input projection + ReLU ----
    const int IND[3] = { 32, 16, 16 };
    for (int t = 0; t < 3; ++t)
        gemm(x_in[t], lin_w[t], lin_b[t], h0[t], NT[t], IND[t], 1);

    // ---- relation metadata: {E, src, dst, firstForDst} ----
    const int RE[5]  = { 1000000, 250000, 250000, 250000, 250000 };
    const int RS[5]  = { 0, 1, 2, 0, 0 };
    const int RD[5]  = { 0, 0, 0, 1, 2 };
    const int RF[5]  = { 1, 0, 0, 1, 1 };

    float** cur = h0;
    float** nxt = h1;
    for (int l = 0; l < 2; ++l) {
        // K/Q/V projections per node type
        for (int t = 0; t < 3; ++t) {
            gemm(cur[t], cv[l].kw[t], cv[l].kb[t], kf[t], NT[t], 64, 0);
            gemm(cur[t], cv[l].qw[t], cv[l].qb[t], qf[t], NT[t], 64, 0);
            gemm(cur[t], cv[l].vw[t], cv[l].vb[t], vf[t], NT[t], 64, 0);
        }
        // per-relation attention + aggregation
        for (int r = 0; r < 5; ++r) {
            int Er = RE[r], src = RS[r], dst = RD[r];
            int Md = NT[dst], Ms = NT[src];
            reltf(kf[src], cv[l].arel[r], krel, Ms);
            reltf(vf[src], cv[l].mrel[r], vrel, Ms);
            {   // init segment max to key(-inf), segment sum to 0
                size_t n = (size_t)Md * 2;
                int g = (int)((n + 255) / 256); if (g > 2048) g = 2048;
                fill_u32_k<<<g,256,0,stream>>>(segmax, n, 0x007FFFFFu);
                zerof(segsum, n);
            }
            edge_score_k<<<(Er + 255) / 256, 256, 0, stream>>>(
                ei[r], Er, qf[dst], krel, cv[l].prel[r], score, segmax);
            edge_expsum_k<<<(Er * 2 + 255) / 256, 256, 0, stream>>>(
                ei[r], Er, score, segmax, segsum);
            float* tgt = RF[r] ? accA[dst] : acctmp;
            zerof(tgt, (size_t)Md * 64);
            edge_message_k<<<(Er * 4 + 255) / 256, 256, 0, stream>>>(
                ei[r], Er, score, segsum, vrel, tgt);
            if (!RF[r]) {
                size_t n = (size_t)Md * 64;
                emin_k<<<(int)((n + 255) / 256), 256, 0, stream>>>(accA[dst], acctmp, n);
            }
        }
        // out = relu( sk*(gelu(acc) @ a_w + a_b) + (1-sk)*h )
        for (int t = 0; t < 3; ++t)
            gemm(accA[t], cv[l].aw[t], cv[l].ab[t], nxt[t], NT[t], 64, 2,
                 cur[t], cv[l].skip[t]);
        float** tmp = cur; cur = nxt; nxt = tmp;
    }

    // ---- MLP on bus nodes: (GEMM -> BN -> ReLU) x2 -> GEMV ----
    gemm(cur[0], mw1, mb1, t1, NBUS, 64, 0);
    zerof(bnS, 64); zerof(bnQ, 64);
    bn_stats_k<<<512,256,0,stream>>>(t1, bnS, bnQ, NBUS);
    bn_apply_k<<<(NBUS * 64 + 255) / 256, 256, 0, stream>>>(t1, bnS, bnQ, mg1, me1, NBUS);

    gemm(t1, mw2, mb2, t2, NBUS, 64, 0);
    zerof(bnS, 64); zerof(bnQ, 64);
    bn_stats_k<<<512,256,0,stream>>>(t2, bnS, bnQ, NBUS);
    bn_apply_k<<<(NBUS * 64 + 255) / 256, 256, 0, stream>>>(t2, bnS, bnQ, mg2, me2, NBUS);

    final_dot_k<<<(NBUS + 255) / 256, 256, 0, stream>>>(t2, mw3, mb3,
                                                        (float*)d_out, NBUS);
}
